// MultiHeadAttention_75548474736720
// MI455X (gfx1250) — matrix operand order repaired and checked by reading, not compiled, not run on hardware
//
#include <hip/hip_runtime.h>
#include <hip/hip_bf16.h>
#include <cstdint>

typedef __bf16 bf16_t;
typedef __attribute__((ext_vector_type(16))) __bf16 v16bf;
typedef __attribute__((ext_vector_type(8)))  float  v8f;

#define HH 16
#define SS 1024
#define DDIM 64

// async global->LDS copy, 16B per lane; INST_OFFSET applies to BOTH addresses
#define ASYNC_B128_OFF(ldsoff, gaddr, IMM)                                     \
  asm volatile("global_load_async_to_lds_b128 %0, %1, off offset:" #IMM        \
               :: "v"(ldsoff), "v"(gaddr) : "memory")

// ---------------------------------------------------------------------------
// Async double-buffered bf16 WMMA GEMM:  D = epi( scale * (A x B^T + bias) )
//   A: [M,K] bf16 row-major;  B: [N,K] bf16 row-major (always transposed form)
//   Batched over blockIdx.z; z -> (bz=z/H2, hz=z%H2); elem offsets bz*s?b+hz*s?h
//   EPI:    0 = none, 1 = elu(x)+1, 2 = causal: x * (row>=col ? 1/(row+1) : 0)
//   SPLITM: 0 = plain [M,ldd] store; 1 = split-head [B,H,S,D]; 2 = [B,H,D,S]
//   TSTORE: also store x transposed to DT[col][row] (square tiles)
// Block: 256 threads (8 wave32), 128x128 tile; wave = 64x32 = 8 wmma tiles.
// Steady state per wave-tile: 4 async b128 issues (next tile), 12 ds_load_b128,
// 8 v_wmma_f32_16x16x32_bf16, s_wait_asynccnt pipelining.
// ---------------------------------------------------------------------------
template<int EPI, int SPLITM, bool OUT_BF16, bool TSTORE>
__global__ __launch_bounds__(256)
void gemm_wmma(const bf16_t* Ap, const bf16_t* Bp, void* Dp, void* DTp,
               const float* bias,
               long sAb, long sAh, long sBb, long sBh, long sDb, long sDh,
               int lda, int ldb, int ldd, int M, int N, int K,
               float scale, int H2)
{
  constexpr int LDT   = 40;                     // lds row stride (80B, 16B aligned)
  constexpr int TILE  = 128 * LDT;
  __shared__ __align__(16) bf16_t sA[2 * TILE];
  __shared__ __align__(16) bf16_t sB[2 * TILE];

  const int tid     = threadIdx.x;
  const int wv      = tid >> 5;
  const int lane    = tid & 31;
  const int lanemod = lane & 15;
  const int hf      = lane >> 4;                // lane half (0/1)

  const int z  = blockIdx.z;
  const int bz = z / H2, hz = z % H2;

  const int rowBase = blockIdx.y * 128;
  const int colBase = blockIdx.x * 128;
  const int waveM   = wv >> 2;                  // 0..1 -> 64 rows
  const int waveN   = wv & 3;                   // 0..3 -> 32 cols

  const bf16_t* A = Ap + (size_t)bz * sAb + (size_t)hz * sAh;
  const bf16_t* B = Bp + (size_t)bz * sBb + (size_t)hz * sBh;

  v8f acc[4][2];
  #pragma unroll
  for (int mi = 0; mi < 4; ++mi)
    #pragma unroll
    for (int ni = 0; ni < 2; ++ni)
      #pragma unroll
      for (int r = 0; r < 8; ++r) acc[mi][ni][r] = 0.0f;

  // ---- issue one 128x32 tile pair as 4 async b128 ops per wave ----
  auto issue = [&](int buf, int kk2) {
    const int r = tid >> 1, c = tid & 1;
    { // A tile: row rowBase+r (always < M: M is a multiple of 128)
      const bf16_t* g = A + (size_t)(rowBase + r) * lda + kk2 + c * 8;
      const uint32_t l = (uint32_t)(uintptr_t)&sA[buf * TILE + r * LDT + c * 8];
      ASYNC_B128_OFF(l, g, 0);
      ASYNC_B128_OFF(l, g, 32);
    }
    { // B tile: row n = colBase+r of B[N,K]; clamp keeps OOB rows in-buffer
      int gn = colBase + r; if (gn >= N) gn = N - 1;
      const bf16_t* g = B + (size_t)gn * ldb + kk2 + c * 8;
      const uint32_t l = (uint32_t)(uintptr_t)&sB[buf * TILE + r * LDT + c * 8];
      ASYNC_B128_OFF(l, g, 0);
      ASYNC_B128_OFF(l, g, 32);
    }
  };

  // ---- fragment build (ISA 7.12.2 layouts) + 8x WMMA on one LDS tile ----
  auto compute = [&](int base) {
    union Frag { v16bf v; uint32_t u[8]; };
    Frag af[4], bfr[2];
    #pragma unroll
    for (int mi = 0; mi < 4; ++mi) {
      const int mrow = base + (waveM * 64 + mi * 16 + lanemod) * LDT;
      #pragma unroll
      for (int vi = 0; vi < 8; ++vi) {
        const int kk = ((vi & 3) << 1) + (hf << 3) + ((vi >> 2) << 4); // V0-3->K0..7, V4-7->K16..23 (+8/half)
        af[mi].u[vi] = *(const uint32_t*)&sA[mrow + kk];
      }
    }
    #pragma unroll
    for (int ni = 0; ni < 2; ++ni) {
      const int nrow = base + (waveN * 32 + ni * 16 + lanemod) * LDT;
      #pragma unroll
      for (int vi = 0; vi < 8; ++vi) {
        const int kk = ((vi & 3) << 1) + (hf << 3) + ((vi >> 2) << 4); // V0-3->K0..7, V4-7->K16..23 (+8/half)
        bfr[ni].u[vi] = *(const uint32_t*)&sB[nrow + kk];
      }
    }
    #pragma unroll
    for (int mi = 0; mi < 4; ++mi)
      #pragma unroll
      for (int ni = 0; ni < 2; ++ni)
        acc[mi][ni] = __builtin_amdgcn_wmma_f32_16x16x32_bf16(
            false, af[mi].v, false, bfr[ni].v, (short)0, acc[mi][ni], false, false);
  };

  // ---- double-buffered async pipeline ----
  issue(0, 0);                                   // 4 async ops / wave
  for (int kk = 0; kk < K; kk += 32) {
    const int  cur  = (kk >> 5) & 1;
    const bool more = (kk + 32) < K;
    if (more) issue(cur ^ 1, kk + 32);           // prefetch next tile (+4 ops)
    if (more) asm volatile("s_wait_asynccnt 0x4" ::: "memory");
    else      asm volatile("s_wait_asynccnt 0x0" ::: "memory");
    __syncthreads();                             // tile visible to all waves
    compute(cur * TILE);
    __syncthreads();                             // done reading before overwrite
  }

  // ---- epilogue + store (C/D: VGPR r -> M=r(+8 for lane half), lane -> N) ----
  #pragma unroll
  for (int mi = 0; mi < 4; ++mi) {
    #pragma unroll
    for (int ni = 0; ni < 2; ++ni) {
      const int mrow0 = rowBase + waveM * 64 + mi * 16 + hf * 8;
      const int ncol  = colBase + waveN * 32 + ni * 16 + lanemod;
      #pragma unroll
      for (int r = 0; r < 8; ++r) {
        const int row = mrow0 + r;
        if (row < M && ncol < N) {
          float x = acc[mi][ni][r];
          if (bias) x += bias[ncol];
          x *= scale;
          if constexpr (EPI == 1) x = (x > 0.0f) ? (x + 1.0f) : __expf(x);  // elu(x)+1
          if constexpr (EPI == 2) x = (row >= ncol) ? x * (1.0f / (float)(row + 1)) : 0.0f;
          if constexpr (OUT_BF16) {
            bf16_t* D = (bf16_t*)Dp + (size_t)bz * sDb + (size_t)hz * sDh;
            if constexpr (SPLITM == 1) {        // [B,H,S,D]
              const int b = row >> 10, t = row & 1023, h = ncol >> 6, d = ncol & 63;
              D[(((size_t)(b * HH + h)) * SS + t) * DDIM + d] = (bf16_t)x;
            } else if constexpr (SPLITM == 2) { // [B,H,D,S]
              const int b = row >> 10, t = row & 1023, h = ncol >> 6, d = ncol & 63;
              D[(((size_t)(b * HH + h)) * DDIM + d) * SS + t] = (bf16_t)x;
            } else {
              D[(size_t)row * ldd + ncol] = (bf16_t)x;
            }
            if constexpr (TSTORE) {             // transposed copy (square)
              bf16_t* DT = (bf16_t*)DTp + (size_t)bz * sDb + (size_t)hz * sDh;
              DT[(size_t)ncol * ldd + row] = (bf16_t)x;
            }
          } else {
            float* D = (float*)Dp + (size_t)bz * sDb + (size_t)hz * sDh;
            D[(size_t)row * ldd + ncol] = x;
          }
        }
      }
    }
  }
}

// ---------------------------------------------------------------------------
// Row softmax over S=1024 columns; one block (256 thr) per row. f32 in, bf16 out.
// ---------------------------------------------------------------------------
__global__ __launch_bounds__(256)
void softmax_rows(const float* __restrict__ X, bf16_t* __restrict__ Y)
{
  const int row = blockIdx.x;
  const int z   = blockIdx.y;
  const float* x = X + ((size_t)z * SS + row) * SS;
  bf16_t*      y = Y + ((size_t)z * SS + row) * SS;
  const int tid = threadIdx.x, lane = tid & 31, wv = tid >> 5;

  __shared__ float red[8];
  __shared__ float bcast;

  float vv[4];
  float mx = -3.4e38f;
  #pragma unroll
  for (int i = 0; i < 4; ++i) { vv[i] = x[tid + i * 256]; mx = fmaxf(mx, vv[i]); }
  #pragma unroll
  for (int o = 16; o; o >>= 1) mx = fmaxf(mx, __shfl_xor(mx, o, 32));
  if (lane == 0) red[wv] = mx;
  __syncthreads();
  if (tid == 0) { float m = red[0];
    #pragma unroll
    for (int k = 1; k < 8; ++k) m = fmaxf(m, red[k]);
    bcast = m; }
  __syncthreads();
  mx = bcast;

  float s = 0.0f;
  #pragma unroll
  for (int i = 0; i < 4; ++i) { vv[i] = __expf(vv[i] - mx); s += vv[i]; }
  #pragma unroll
  for (int o = 16; o; o >>= 1) s += __shfl_xor(s, o, 32);
  __syncthreads();
  if (lane == 0) red[wv] = s;
  __syncthreads();
  if (tid == 0) { float t = 0.0f;
    #pragma unroll
    for (int k = 0; k < 8; ++k) t += red[k];
    bcast = t; }
  __syncthreads();
  const float inv = 1.0f / bcast;
  #pragma unroll
  for (int i = 0; i < 4; ++i) y[tid + i * 256] = (bf16_t)(vv[i] * inv);
}

// f32 -> bf16 elementwise
__global__ __launch_bounds__(256)
void convert_bf16(const float* __restrict__ X, bf16_t* __restrict__ Y)
{
  const size_t i = (size_t)blockIdx.x * 256 + threadIdx.x;
  Y[i] = (bf16_t)X[i];
}

// W f32 [1024,1024] -> WT bf16 [1024,1024] transposed
__global__ __launch_bounds__(256)
void transpose_bf16(const float* __restrict__ W, bf16_t* __restrict__ WT)
{
  const size_t i = (size_t)blockIdx.x * 256 + threadIdx.x;   // < 1M
  const int r = (int)(i >> 10), c = (int)(i & 1023);
  WT[((size_t)c << 10) + r] = (bf16_t)W[i];
}

// p (f32 [B,S,Dm]) -> ph (bf16 [B,H,S,D])
__global__ __launch_bounds__(256)
void split_heads_bf16(const float* __restrict__ P, bf16_t* __restrict__ Ph)
{
  const size_t i = (size_t)blockIdx.x * 256 + threadIdx.x;   // < 2*1024*1024
  const int b = (int)(i >> 20);
  const int rem = (int)(i & ((1u << 20) - 1));
  const int t = rem >> 10;
  const int col = rem & 1023;
  const int h = col >> 6, d = col & 63;
  Ph[(((size_t)(b * HH + h)) * SS + t) * DDIM + d] = (bf16_t)P[i];
}

extern "C" void kernel_launch(void* const* d_in, const int* in_sizes, int n_in,
                              void* d_out, int out_size, void* d_ws, size_t ws_size,
                              hipStream_t stream) {
  (void)in_sizes; (void)n_in; (void)out_size; (void)ws_size;
  const float* v    = (const float*)d_in[0];
  const float* k    = (const float*)d_in[1];
  const float* q    = (const float*)d_in[2];
  const float* p    = (const float*)d_in[3];
  const float* wq_k = (const float*)d_in[4];
  const float* wq_b = (const float*)d_in[5];
  const float* wk_k = (const float*)d_in[6];
  const float* wk_b = (const float*)d_in[7];
  const float* wv_k = (const float*)d_in[8];
  const float* wv_b = (const float*)d_in[9];
  const float* wc_k = (const float*)d_in[10];
  const float* wc_b = (const float*)d_in[11];
  float* out = (float*)d_out;

  // workspace carve-up (256B aligned)
  char* ws = (char*)d_ws;
  size_t off = 0;
  auto take = [&](size_t bytes) -> void* {
    void* ptr = ws + off;
    off += (bytes + 255) & ~(size_t)255;
    return ptr;
  };
  const size_t NBH = 2ull * HH;                    // 32 batched heads
  const size_t SD  = (size_t)SS * DDIM;            // 65536
  const size_t SSx = (size_t)SS * SS;              // 1M
  const size_t DM2 = 2048ull * 1024;               // B*S x Dm
  const size_t WW  = 1024ull * 1024;

  bf16_t* qb    = (bf16_t*)take(DM2 * 2);          // bf16 copies of q,k,v
  bf16_t* kb    = (bf16_t*)take(DM2 * 2);
  bf16_t* vb    = (bf16_t*)take(DM2 * 2);
  bf16_t* wqT   = (bf16_t*)take(WW * 2);           // transposed bf16 weights
  bf16_t* wkT   = (bf16_t*)take(WW * 2);
  bf16_t* wvT   = (bf16_t*)take(WW * 2);
  bf16_t* wcT   = (bf16_t*)take(WW * 2);
  bf16_t* qh    = (bf16_t*)take(NBH * SD * 2);     // [B,H,S,D]
  bf16_t* kh    = (bf16_t*)take(NBH * SD * 2);     // [B,H,S,D]
  bf16_t* vhT   = (bf16_t*)take(NBH * SD * 2);     // [B,H,D,S]
  bf16_t* ph    = (bf16_t*)take(NBH * SD * 2);     // [B,H,S,D]
  bf16_t* pack  = (bf16_t*)take(NBH * SSx * 2);    // [B,H,S,S]
  bf16_t* packT = (bf16_t*)take(NBH * SSx * 2);    // transposed per head
  bf16_t* s2    = (bf16_t*)take(NBH * SSx * 2);    // scores2, reused as softmax out
  float*  unk   = (float*) take(NBH * SSx * 4);    // pre-softmax f32
  bf16_t* s3    = (bf16_t*)unk;                    // reuse unk region after softmax
  bf16_t* ao    = (bf16_t*)take(DM2 * 2);          // merged [B,S,Dm] bf16

  dim3 blk(256);
  const float norm_d = 0.125f;                     // 1/sqrt(64)

  // 0: precision/layout pre-passes (tiny traffic)
  convert_bf16<<<8192, blk, 0, stream>>>(q, qb);
  convert_bf16<<<8192, blk, 0, stream>>>(k, kb);
  convert_bf16<<<8192, blk, 0, stream>>>(v, vb);
  transpose_bf16<<<4096, blk, 0, stream>>>(wq_k, wqT);
  transpose_bf16<<<4096, blk, 0, stream>>>(wk_k, wkT);
  transpose_bf16<<<4096, blk, 0, stream>>>(wv_k, wvT);
  transpose_bf16<<<4096, blk, 0, stream>>>(wc_k, wcT);
  split_heads_bf16<<<8192, blk, 0, stream>>>(p, ph);

  // 1-3: projections  (A @ WT^T + b) * scale -> split-head layouts
  gemm_wmma<0,1,true ,false><<<dim3(8,16,1), blk, 0, stream>>>(
      qb, wqT, qh, nullptr, wq_b, 0,0, 0,0, 0,0,
      1024,1024,0, 2048,1024,1024, norm_d, 1);
  gemm_wmma<0,1,true ,false><<<dim3(8,16,1), blk, 0, stream>>>(
      kb, wkT, kh, nullptr, wk_b, 0,0, 0,0, 0,0,
      1024,1024,0, 2048,1024,1024, 1.0f, 1);
  gemm_wmma<0,2,true ,false><<<dim3(8,16,1), blk, 0, stream>>>(
      vb, wvT, vhT, nullptr, wv_b, 0,0, 0,0, 0,0,
      1024,1024,0, 2048,1024,1024, 1.0f, 1);

  // 4: pack[t,s] = elu(qh[t].ph[s])+1, also store packT  (K=64)
  gemm_wmma<1,0,true ,true ><<<dim3(8,8,32), blk, 0, stream>>>(
      qh, ph, pack, packT, nullptr,
      (long)(HH*SD),(long)SD, (long)(HH*SD),(long)SD, (long)(HH*SSx),(long)SSx,
      64,64,1024, 1024,1024,64, 1.0f, HH);

  // 5: s2[t,s] = causal( qh[t].kh[s] )  (K=64)
  gemm_wmma<2,0,true ,false><<<dim3(8,8,32), blk, 0, stream>>>(
      qh, kh, s2, nullptr, nullptr,
      (long)(HH*SD),(long)SD, (long)(HH*SD),(long)SD, (long)(HH*SSx),(long)SSx,
      64,64,1024, 1024,1024,64, 1.0f, HH);

  // 6: unk[t,j] = sum_s s2[t,s]*pack[s,j] = s2 @ packT^T   (1024^3/head, f32)
  gemm_wmma<0,0,false,false><<<dim3(8,8,32), blk, 0, stream>>>(
      s2, packT, unk, nullptr, nullptr,
      (long)(HH*SSx),(long)SSx, (long)(HH*SSx),(long)SSx, (long)(HH*SSx),(long)SSx,
      1024,1024,1024, 1024,1024,1024, 1.0f, HH);

  // 7: softmax rows of unk -> s2 (bf16, reused)
  softmax_rows<<<dim3(1024, 32), blk, 0, stream>>>(unk, s2);

  // 8: s3[t,s] = causal( soft[t] . pack[s] )   (1024^3/head)
  gemm_wmma<2,0,true ,false><<<dim3(8,8,32), blk, 0, stream>>>(
      s2, pack, s3, nullptr, nullptr,
      (long)(HH*SSx),(long)SSx, (long)(HH*SSx),(long)SSx, (long)(HH*SSx),(long)SSx,
      1024,1024,1024, 1024,1024,1024, 1.0f, HH);

  // 9: ao[t,d] = sum_s s3[t,s]*vh[s,d] = s3 @ vhT^T, merged [B,S,Dm] store
  gemm_wmma<0,0,true ,false><<<dim3(1,8,32), blk, 0, stream>>>(
      s3, vhT, ao, nullptr, nullptr,
      (long)(HH*SSx),(long)SSx, (long)(HH*SD),(long)SD,
      (long)((size_t)SS*1024), (long)DDIM,
      1024,1024,1024, 1024,64,1024, 1.0f, HH);

  // 10: out = ao @ wcT^T + wc_b   (f32 out)
  gemm_wmma<0,0,false,false><<<dim3(8,16,1), blk, 0, stream>>>(
      ao, wcT, out, nullptr, wc_b, 0,0, 0,0, 0,0,
      1024,1024,1024, 2048,1024,1024, 1.0f, 1);
}
